// RSGCN_76141180223726
// MI455X (gfx1250) — matrix operand order repaired and checked by reading, not compiled, not run on hardware
//
#include <hip/hip_runtime.h>
#include <hip/hip_bf16.h>
#include <math.h>

#define NNODE 50000
#define DIN   768
#define NHID  256
#define BM    80      // 5 M-tiles of 16;  50000 = 625 * 80 exactly
#define BN    384     // 8 waves x 3 N-tiles of 16; 768 = 2 * 384
#define BK    32
#define NK    (DIN / BK)   // 24 K-steps
#define LSA_ELEMS (BM * BK)   // 2560 bf16 = 5 KB
#define LSB_ELEMS (BN * BK)   // 12288 bf16 = 24 KB

typedef __attribute__((ext_vector_type(16))) __bf16 v16bf;
typedef __attribute__((ext_vector_type(8)))  __bf16 v8bf;
typedef __attribute__((ext_vector_type(4)))  __bf16 v4bf;
typedef __attribute__((ext_vector_type(8)))  float  v8f;

#if defined(__AMDGCN__) && __has_builtin(__builtin_amdgcn_tensor_load_to_lds) && \
    __has_builtin(__builtin_amdgcn_s_wait_tensorcnt)
#define USE_TDM 1
#else
#define USE_TDM 0
#endif

__device__ __forceinline__ v16bf frag16(const __bf16* p0, const __bf16* p1) {
    v8bf lo = *(const v8bf*)p0;
    v8bf hi = *(const v8bf*)p1;
    return __builtin_shufflevector(lo, hi, 0,1,2,3,4,5,6,7,8,9,10,11,12,13,14,15);
}

#if USE_TDM
typedef __attribute__((ext_vector_type(4))) unsigned v4u;
typedef __attribute__((ext_vector_type(8))) int      v8i;
typedef __attribute__((ext_vector_type(4))) int      v4i;

// Issue one TDM 2-D tile load: tileRows rows of tileK bf16 elements, row
// stride rowStride elements, packed contiguously into LDS at ldsByte.
// D# layout per CDNA5 ISA 8.3/8.4: group0={count=1, lds_addr, global_addr,
// type=2}; group1={data_size=1(2B), dims, tile dims, dim0 stride}.
static __device__ __forceinline__ void tdm_load_tile(
    unsigned ldsByte, const __bf16* gsrc, unsigned tileK, unsigned tileRows,
    unsigned rowStride)
{
    unsigned long long ga = (unsigned long long)(uintptr_t)gsrc;
    v4u g0 = { 1u,                                   // count=1, user desc
               ldsByte,                              // lds_addr
               (unsigned)ga,                         // global_addr[31:0]
               (unsigned)((ga >> 32) & 0x01FFFFFFull) | (2u << 30) }; // +type=2
    unsigned dim0 = rowStride;                       // tensor width (elems)
    unsigned dim1 = 1u << 20;                        // generous row bound
    v8i g1 = { (int)0x00010000u,                                   // data_size=2B
               (int)((dim0 & 0xFFFFu) << 16),                      // dim0 lo
               (int)(((dim0 >> 16) & 0xFFFFu) | ((dim1 & 0xFFFFu) << 16)),
               (int)(((dim1 >> 16) & 0xFFFFu) | (tileK << 16)),    // tile_dim0
               (int)tileRows,                                      // tile_dim1
               (int)rowStride,                                     // dim0_stride
               0, 0 };
    v4i z4 = { 0, 0, 0, 0 };
#if __clang_major__ >= 23
    v8i z8 = { 0, 0, 0, 0, 0, 0, 0, 0 };
    __builtin_amdgcn_tensor_load_to_lds(g0, g1, z4, z4, z8, 0);
#else
    __builtin_amdgcn_tensor_load_to_lds(g0, g1, z4, z4, 0);
#endif
}
#endif

// ---------------------------------------------------------------------------
// Prep: x (fp32, row-major [m][k]) -> bf16, vectorized 4-wide.
// ---------------------------------------------------------------------------
__global__ __launch_bounds__(256) void cvt_x(const float* __restrict__ x,
                                             __bf16* __restrict__ xb, int n4) {
    int i = blockIdx.x * 256 + threadIdx.x;
    if (i < n4) {
        float4 f = ((const float4*)x)[i];
        v4bf p = { (__bf16)f.x, (__bf16)f.y, (__bf16)f.z, (__bf16)f.w };
        *(v4bf*)&xb[(size_t)i * 4] = p;
    }
}

// Prep: WT[n][k] = W_{n>>8}[k][n&255] as bf16 (fused-N transposed weights).
__global__ __launch_bounds__(256) void build_wt(
    const float* __restrict__ W0, const float* __restrict__ W1,
    const float* __restrict__ W2, __bf16* __restrict__ WT) {
    int n = blockIdx.x;
    int z = n >> 8, c = n & 255;
    const float* W = (z == 0) ? W0 : ((z == 1) ? W1 : W2);
    for (int k = threadIdx.x; k < DIN; k += 256)
        WT[(size_t)n * DIN + k] = (__bf16)W[(size_t)k * NHID + c];
}

// ---------------------------------------------------------------------------
// Layer-1 fused triple GEMM over concatenated N=768.  bf16 WMMA, f32 acc,
// double-buffered LDS staged by the Tensor Data Mover (wave 0 issues,
// TENSORcnt-waited).  Since the TDM writes LDS behind the compiler's back,
// the shared arena is escaped into a volatile asm and a memory-clobber asm
// fences every K-step so fragment ds_loads are neither undef-folded nor
// CSE'd across buffer reuse.  Plain-copy fallback if the builtin is absent.
// Grid: (625, 2), block 256 (8 waves); 15 WMMA / wave / K-step.
// ---------------------------------------------------------------------------
__global__ __launch_bounds__(256) void gemm_l1(
    const __bf16* __restrict__ xb,   // [50000][768] bf16
    const __bf16* __restrict__ WT,   // [768 n][768 k] bf16
    const float* __restrict__ mlp1b,
    float* __restrict__ o0, float* __restrict__ o1, float* __restrict__ o2)
{
    // Single arena so LDS byte offsets are statically known (group seg base 0).
    __shared__ __align__(16) __bf16 smem[2 * LSA_ELEMS + 2 * LSB_ELEMS]; // 58 KB
    __bf16* lsA0 = smem;
    __bf16* lsA1 = smem + LSA_ELEMS;
    __bf16* lsB0 = smem + 2 * LSA_ELEMS;
    __bf16* lsB1 = smem + 2 * LSA_ELEMS + LSB_ELEMS;

    const int tid   = threadIdx.x;
    const int wave  = tid >> 5;
    const int lane  = tid & 31;
    const int mBase = blockIdx.x * BM;
    const int nBase = blockIdx.y * BN;

#if !USE_TDM
    auto stageA = [&](int kB, __bf16* dst) {
        #pragma unroll
        for (int it = 0; it < 2; ++it) {
            int i = tid + it * 256;             // 320 chunks of 8 bf16
            if (i < LSA_ELEMS / 8) {
                int r = i >> 2, c = (i & 3) * 8;
                *(v8bf*)&dst[r * BK + c] =
                    *(const v8bf*)&xb[(size_t)(mBase + r) * DIN + kB + c];
            }
        }
    };
    auto stageB = [&](int kB, __bf16* dst) {
        #pragma unroll
        for (int it = 0; it < 6; ++it) {
            int i = tid + it * 256;             // 1536 chunks, exact
            int n = i >> 2, c = (i & 3) * 8;
            *(v8bf*)&dst[n * BK + c] =
                *(const v8bf*)&WT[(size_t)(nBase + n) * DIN + kB + c];
        }
    };
#endif

    v8f acc[3][5] = {};

#if USE_TDM
    if (wave == 0) {
        tdm_load_tile(0u, &xb[(size_t)mBase * DIN], BK, BM, DIN);
        tdm_load_tile(4u * LSA_ELEMS, &WT[(size_t)nBase * DIN], BK, BN, DIN);
        __builtin_amdgcn_s_wait_tensorcnt((short)0);
    }
    // Escape the LDS arena: TDM is an invisible writer; after this, any
    // memory-clobber asm is a potential store to smem.
    asm volatile("" :: "v"(smem) : "memory");
#else
    stageA(0, lsA0);
    stageB(0, lsB0);
#endif
    __syncthreads();

    const int kb = (lane >> 4) * 16;   // A-frag K base (contiguous 16)
    const int ks = (lane >> 4) * 16;   // B-frag K base (contiguous 16)

    for (int kstep = 0; kstep < NK; ++kstep) {
#if USE_TDM
        // Per-iteration AA fence: LDS contents may have changed (TDM).
        asm volatile("" :: "v"(smem) : "memory");
#endif
        int buf = kstep & 1;
        __bf16* curA = buf ? lsA1 : lsA0;
        __bf16* curB = buf ? lsB1 : lsB0;

        if (kstep + 1 < NK) {          // stage next tile into the other buffer
            int kB = (kstep + 1) * BK;
#if USE_TDM
            if (wave == 0) {
                unsigned nb = buf ^ 1;
                tdm_load_tile(nb * (2u * LSA_ELEMS),
                              &xb[(size_t)mBase * DIN + kB], BK, BM, DIN);
                tdm_load_tile(4u * LSA_ELEMS + nb * (2u * LSB_ELEMS),
                              &WT[(size_t)nBase * DIN + kB], BK, BN, DIN);
            }
#else
            stageA(kB, buf ? lsA0 : lsA1);
            stageB(kB, buf ? lsB0 : lsB1);
#endif
        }
        if (kstep + 2 < NK)            // pull K+2 toward L2 (global_prefetch_b8)
            __builtin_prefetch(&xb[(size_t)(mBase + (tid >> 2)) * DIN + (kstep + 2) * BK], 0, 1);

        v16bf af[5];
        #pragma unroll
        for (int t = 0; t < 5; ++t) {
            const __bf16* ap = &curA[(t * 16 + (lane & 15)) * BK + kb];
            af[t] = frag16(ap, ap + 8);
        }
        #pragma unroll
        for (int j = 0; j < 3; ++j) {
            const __bf16* bp = &curB[(wave * 48 + j * 16 + (lane & 15)) * BK + ks];
            v16bf bf = frag16(bp, bp + 8);
            #pragma unroll
            for (int t = 0; t < 5; ++t)
                acc[j][t] = __builtin_amdgcn_wmma_f32_16x16x32_bf16(
                    false, af[t], false, bf, (short)0, acc[j][t], false, false);
        }
#if USE_TDM
        if (wave == 0) __builtin_amdgcn_s_wait_tensorcnt((short)0);
#endif
        __syncthreads();
    }

    // Epilogue: C/D layout — VGPR r: M = r + (lane>=16 ? 8 : 0), N = lane&15.
    #pragma unroll
    for (int j = 0; j < 3; ++j) {
        int col = nBase + wave * 48 + j * 16 + (lane & 15);
        int z = col >> 8, c = col & 255;
        float* out = (z == 0) ? o0 : ((z == 1) ? o1 : o2);
        float bv = (z == 2) ? mlp1b[c] : 0.0f;
        #pragma unroll
        for (int t = 0; t < 5; ++t) {
            int row0 = mBase + t * 16 + ((lane >> 4) * 8);
            #pragma unroll
            for (int r = 0; r < 8; ++r) {
                float v = acc[j][t][r];
                if (z == 2) v = tanhf(v + bv);
                out[(size_t)(row0 + r) * NHID + c] = v;
            }
        }
    }
}

// Initialize SpMM output with its bias (reference: spmm(...) + bias).
__global__ __launch_bounds__(256) void bias_init256(float* __restrict__ out,
                                                    const float* __restrict__ bias) {
    out[(size_t)blockIdx.x * NHID + threadIdx.x] = bias[threadIdx.x];
}

// Edge-parallel SpMM (256 cols): block handles 16 edges, thread = one column.
__global__ __launch_bounds__(256) void spmm_h(
    const int* __restrict__ rows, const int* __restrict__ cols,
    const float* __restrict__ vals,
    const float* __restrict__ H, float* __restrict__ out, int E)
{
    int base = blockIdx.x * 16;
    int c = threadIdx.x;
    #pragma unroll
    for (int e = 0; e < 16; ++e) {
        int idx = base + e;
        if (idx >= E) return;
        int r = rows[idx], cl = cols[idx];
        float v = vals[idx];
        atomicAdd(&out[(size_t)r * NHID + c], v * H[(size_t)cl * NHID + c]);
    }
}

// Attention fusion: one wave32 per node; 3 concat-dots, softmax over 3, blend.
__global__ __launch_bounds__(256) void att_fuse(
    const float* __restrict__ R1, const float* __restrict__ R2,
    const float* __restrict__ T3, const float* __restrict__ afw,
    float* __restrict__ X1)
{
    int node = blockIdx.x * 8 + (threadIdx.x >> 5);
    int l = threadIdx.x & 31;
    float s0 = 0.f, s1 = 0.f, s2 = 0.f;
    for (int h = l; h < NHID; h += 32) {
        float a = R1[(size_t)node * NHID + h];
        float b = R2[(size_t)node * NHID + h];
        float c = T3[(size_t)node * NHID + h];
        s0 += a * afw[h * 3 + 0] + b * afw[(NHID + h) * 3 + 0] + c * afw[(2 * NHID + h) * 3 + 0];
        s1 += a * afw[h * 3 + 1] + b * afw[(NHID + h) * 3 + 1] + c * afw[(2 * NHID + h) * 3 + 1];
        s2 += a * afw[h * 3 + 2] + b * afw[(NHID + h) * 3 + 2] + c * afw[(2 * NHID + h) * 3 + 2];
    }
    for (int off = 16; off > 0; off >>= 1) {
        s0 += __shfl_xor(s0, off, 32);
        s1 += __shfl_xor(s1, off, 32);
        s2 += __shfl_xor(s2, off, 32);
    }
    float a0 = fabsf(s0), a1 = fabsf(s1), a2 = fabsf(s2);
    float m  = fmaxf(a0, fmaxf(a1, a2));
    float e0 = __expf(a0 - m), e1 = __expf(a1 - m), e2 = __expf(a2 - m);
    float inv = 1.0f / (e0 + e1 + e2);
    e0 *= inv; e1 *= inv; e2 *= inv;
    for (int h = l; h < NHID; h += 32) {
        size_t o = (size_t)node * NHID + h;
        X1[o] = R1[o] * e0 + R2[o] * e1 + T3[o] * e2;
    }
}

// Layer-2: three N=2 GEMV-like projections per node; one wave32 per node.
__global__ __launch_bounds__(256) void layer2_gemm(
    const float* __restrict__ X1,
    const float* __restrict__ W2r1, const float* __restrict__ W2r2,
    const float* __restrict__ mlp2W, const float* __restrict__ mlp2b,
    float* __restrict__ G1, float* __restrict__ G2, float* __restrict__ T2)
{
    int node = blockIdx.x * 8 + (threadIdx.x >> 5);
    int l = threadIdx.x & 31;
    float a0=0.f,a1=0.f,a2=0.f,a3=0.f,a4=0.f,a5=0.f;
    for (int h = l; h < NHID; h += 32) {
        float xv = X1[(size_t)node * NHID + h];
        a0 += xv * W2r1[h * 2 + 0];  a1 += xv * W2r1[h * 2 + 1];
        a2 += xv * W2r2[h * 2 + 0];  a3 += xv * W2r2[h * 2 + 1];
        a4 += xv * mlp2W[h * 2 + 0]; a5 += xv * mlp2W[h * 2 + 1];
    }
    for (int off = 16; off > 0; off >>= 1) {
        a0 += __shfl_xor(a0, off, 32); a1 += __shfl_xor(a1, off, 32);
        a2 += __shfl_xor(a2, off, 32); a3 += __shfl_xor(a3, off, 32);
        a4 += __shfl_xor(a4, off, 32); a5 += __shfl_xor(a5, off, 32);
    }
    if (l == 0) {
        G1[node * 2 + 0] = a0;  G1[node * 2 + 1] = a1;
        G2[node * 2 + 0] = a2;  G2[node * 2 + 1] = a3;
        T2[node * 2 + 0] = tanhf(a4 + mlp2b[0]);
        T2[node * 2 + 1] = tanhf(a5 + mlp2b[1]);
    }
}

// Final output init: out = b2r1 + b2r2 + tanh-branch; SpMMs accumulate on top.
__global__ __launch_bounds__(256) void final_init(
    float* __restrict__ out, const float* __restrict__ b2r1,
    const float* __restrict__ b2r2, const float* __restrict__ T2, int n)
{
    int i = blockIdx.x * 256 + threadIdx.x;
    if (i < n) out[i] = b2r1[i & 1] + b2r2[i & 1] + T2[i];
}

// 2-wide edge-parallel SpMM (one thread per edge).
__global__ __launch_bounds__(256) void spmm2(
    const int* __restrict__ rows, const int* __restrict__ cols,
    const float* __restrict__ vals,
    const float* __restrict__ G, float* __restrict__ out, int E)
{
    int e = blockIdx.x * 256 + threadIdx.x;
    if (e >= E) return;
    int r = rows[e], c = cols[e];
    float v = vals[e];
    atomicAdd(&out[r * 2 + 0], v * G[c * 2 + 0]);
    atomicAdd(&out[r * 2 + 1], v * G[c * 2 + 1]);
}

extern "C" void kernel_launch(void* const* d_in, const int* in_sizes, int n_in,
                              void* d_out, int out_size, void* d_ws, size_t ws_size,
                              hipStream_t stream) {
    const float* x     = (const float*)d_in[0];
    const int*   arows = (const int*)d_in[1];
    const int*   acols = (const int*)d_in[2];
    const float* avals = (const float*)d_in[3];
    const int*   srows = (const int*)d_in[4];
    const int*   scols = (const int*)d_in[5];
    const float* svals = (const float*)d_in[6];
    const float* W1r1  = (const float*)d_in[7];
    const float* b1r1  = (const float*)d_in[8];
    const float* W1r2  = (const float*)d_in[9];
    const float* b1r2  = (const float*)d_in[10];
    const float* mlp1W = (const float*)d_in[11];
    const float* mlp1b = (const float*)d_in[12];
    const float* af1w  = (const float*)d_in[13];
    const float* W2r1  = (const float*)d_in[14];
    const float* b2r1  = (const float*)d_in[15];
    const float* W2r2  = (const float*)d_in[16];
    const float* b2r2  = (const float*)d_in[17];
    const float* mlp2W = (const float*)d_in[18];
    const float* mlp2b = (const float*)d_in[19];
    const int E = in_sizes[1];

    float* ws = (float*)d_ws;
    const size_t NH = (size_t)NNODE * NHID;
    __bf16* xb = (__bf16*)ws;                                 // 38.4M bf16
    __bf16* WT = (__bf16*)(ws + 19200000);                    // 589824 bf16
    float* big = ws + 19200000 + 294912;
    float* H1 = big;            // x @ W1r1
    float* H2 = big + NH;       // x @ W1r2
    float* H3 = big + 2 * NH;   // tanh(x @ mlp1_W + b)
    float* R1 = big + 3 * NH;   // spmm(adj, H1) + b1r1
    float* R2 = big + 4 * NH;   // spmm(s, H2) + b1r2
    float* X1 = H1;             // fused (H1 dead after spmm)
    float* G1 = H2;             // layer-2 branch outputs (H2 dead after fusion)
    float* G2 = H2 + 2 * NNODE;
    float* T2 = H2 + 4 * NNODE;
    float* out = (float*)d_out;

    cvt_x<<<(NNODE * DIN / 4 + 255) / 256, 256, 0, stream>>>(x, xb, NNODE * DIN / 4);
    build_wt<<<DIN, 256, 0, stream>>>(W1r1, W1r2, mlp1W, WT);
    gemm_l1<<<dim3(NNODE / BM, DIN / BN), 256, 0, stream>>>(xb, WT, mlp1b, H1, H2, H3);
    bias_init256<<<NNODE, 256, 0, stream>>>(R1, b1r1);
    bias_init256<<<NNODE, 256, 0, stream>>>(R2, b1r2);
    spmm_h<<<(E + 15) / 16, 256, 0, stream>>>(arows, acols, avals, H1, R1, E);
    spmm_h<<<(E + 15) / 16, 256, 0, stream>>>(srows, scols, svals, H2, R2, E);
    att_fuse<<<NNODE / 8, 256, 0, stream>>>(R1, R2, H3, af1w, X1);
    layer2_gemm<<<NNODE / 8, 256, 0, stream>>>(X1, W2r1, W2r2, mlp2W, mlp2b, G1, G2, T2);
    final_init<<<(NNODE * 2 + 255) / 256, 256, 0, stream>>>(out, b2r1, b2r2, T2, NNODE * 2);
    spmm2<<<(E + 255) / 256, 256, 0, stream>>>(arows, acols, avals, G1, out, E);
    spmm2<<<(E + 255) / 256, 256, 0, stream>>>(srows, scols, svals, G2, out, E);
}
